// BitMGQA_25280177504456
// MI455X (gfx1250) — 16-bit and 8-bit matrix operand order repaired and checked by reading, compiled, not run on hardware
//
#include <hip/hip_runtime.h>
#include <cstdint>
#include <cstddef>


typedef __attribute__((ext_vector_type(16))) _Float16 v16h;
typedef __attribute__((ext_vector_type(8)))  float    v8f;
typedef __attribute__((ext_vector_type(8)))  int      v8i;
typedef int v4i_gv __attribute__((vector_size(16)));

union HFrag { v16h v; _Float16 h[16]; unsigned u[8]; };
union IFrag { v8i  v; unsigned u[8]; };

constexpr int kBt  = 2;
constexpr int kS   = 2048;
constexpr int kT   = kBt * kS;
constexpr int kD   = 1024;
constexpr int kKV  = 256;
constexpr int kHD  = 128;
constexpr int kQH  = 8;

#if __has_builtin(__builtin_amdgcn_global_load_async_to_lds_b128)
#define USE_ASYNC_LDS 1
#else
#define USE_ASYNC_LDS 0
#endif

#define AS1P(p) ((__attribute__((address_space(1))) v4i_gv*)(p))
#define AS3P(p) ((__attribute__((address_space(3))) v4i_gv*)(p))

template <int N>
__device__ __forceinline__ void wait_asynccnt() {
#if USE_ASYNC_LDS
#if __has_builtin(__builtin_amdgcn_s_wait_asynccnt)
    __builtin_amdgcn_s_wait_asynccnt((unsigned short)N);
#else
    asm volatile("s_wait_asynccnt %0" :: "i"(N) : "memory");
#endif
#endif
}

__device__ __forceinline__ void stage_panel(const int8_t* __restrict__ W8,
                                            int8_t* lbase, int ocol0, int kb, int tid) {
#pragma unroll
    for (int i = 0; i < 2; ++i) {
        const int c    = tid * 2 + i;
        const int col  = c >> 2;
        const int boff = (c & 3) * 16;
        const int8_t* g = W8 + (size_t)(ocol0 + col) * kD + kb * 64 + boff;
        int8_t*       l = lbase + col * 64 + boff;
#if USE_ASYNC_LDS
        __builtin_amdgcn_global_load_async_to_lds_b128(AS1P(g), AS3P(l), 0, 0);
#else
        *(uint4*)l = *(const uint4*)g;
#endif
    }
}

__global__ void weight_prep(const float* __restrict__ w, int8_t* __restrict__ sgn,
                            float* __restrict__ partial, int n) {
    const int tid = threadIdx.x;
    float acc = 0.f;
    for (int i = blockIdx.x * blockDim.x + tid; i < n; i += gridDim.x * blockDim.x) {
        float s = w[i];
        sgn[i] = (int8_t)((s > 0.f) ? 1 : ((s < 0.f) ? -1 : 0));
        acc += fabsf(s);
    }
    __shared__ float sb[256];
    sb[tid] = acc;
    __syncthreads();
    for (int o = 128; o > 0; o >>= 1) {
        if (tid < o) sb[tid] += sb[tid + o];
        __syncthreads();
    }
    if (tid == 0) partial[blockIdx.x] = sb[0];
}

__global__ void alpha_reduce(const float* __restrict__ partials, float* __restrict__ alphas) {
    const int tid = threadIdx.x;
    __shared__ float sb[256];
    sb[tid] = partials[blockIdx.x * 256 + tid];
    __syncthreads();
    for (int o = 128; o > 0; o >>= 1) {
        if (tid < o) sb[tid] += sb[tid + o];
        __syncthreads();
    }
    if (tid == 0) {
        float invN = (blockIdx.x == 1 || blockIdx.x == 2) ? (1.f / 262144.f)
                                                          : (1.f / 1048576.f);
        alphas[blockIdx.x] = sb[0] * invN;
    }
}

__global__ void act_quant(const float* __restrict__ x, int8_t* __restrict__ q8,
                          float* __restrict__ invs) {
    const int tid = threadIdx.x;
    const size_t t = blockIdx.x;
    const float* xr = x + t * kD;
    float v[4], ss = 0.f, mx = 0.f;
#pragma unroll
    for (int i = 0; i < 4; ++i) {
        v[i] = xr[tid + i * 256];
        ss += v[i] * v[i];
        mx = fmaxf(mx, fabsf(v[i]));
    }
    __shared__ float s1[256], s2[256];
    s1[tid] = ss; s2[tid] = mx;
    __syncthreads();
    for (int o = 128; o > 0; o >>= 1) {
        if (tid < o) { s1[tid] += s1[tid + o]; s2[tid] = fmaxf(s2[tid], s2[tid + o]); }
        __syncthreads();
    }
    const float rms = sqrtf(s1[0] * (1.f / 1024.f) + 1e-6f);
    const float nf  = 0.03125f / rms;
    const float scale = 127.f / fmaxf(s2[0] * nf, 1e-5f);
    if (tid == 0) invs[t] = 1.f / scale;
#pragma unroll
    for (int i = 0; i < 4; ++i) {
        float q = roundf(v[i] * nf * scale);
        q = fminf(fmaxf(q, -128.f), 127.f);
        q8[t * kD + tid + i * 256] = (int8_t)q;
    }
}

__global__ void ln_act_quant(const float* __restrict__ x, const float* __restrict__ g,
                             const float* __restrict__ bv, int8_t* __restrict__ q8,
                             float* __restrict__ invs) {
    const int tid = threadIdx.x;
    const size_t t = blockIdx.x;
    const float* xr = x + t * kD;
    float v[4], su = 0.f, sq = 0.f;
#pragma unroll
    for (int i = 0; i < 4; ++i) { v[i] = xr[tid + i * 256]; su += v[i]; sq += v[i] * v[i]; }
    __shared__ float s1[256], s2[256];
    s1[tid] = su; s2[tid] = sq;
    __syncthreads();
    for (int o = 128; o > 0; o >>= 1) {
        if (tid < o) { s1[tid] += s1[tid + o]; s2[tid] += s2[tid + o]; }
        __syncthreads();
    }
    const float mu  = s1[0] * (1.f / 1024.f);
    const float var = s2[0] * (1.f / 1024.f) - mu * mu;
    const float is  = rsqrtf(var + 1e-5f);
    __syncthreads();
    float y[4], ss = 0.f, mx = 0.f;
#pragma unroll
    for (int i = 0; i < 4; ++i) {
        const int c = tid + i * 256;
        y[i] = (v[i] - mu) * is * g[c] + bv[c];
        ss += y[i] * y[i];
        mx = fmaxf(mx, fabsf(y[i]));
    }
    s1[tid] = ss; s2[tid] = mx;
    __syncthreads();
    for (int o = 128; o > 0; o >>= 1) {
        if (tid < o) { s1[tid] += s1[tid + o]; s2[tid] = fmaxf(s2[tid], s2[tid + o]); }
        __syncthreads();
    }
    const float rms = sqrtf(s1[0] * (1.f / 1024.f) + 1e-6f);
    const float nf  = 0.03125f / rms;
    const float scale = 127.f / fmaxf(s2[0] * nf, 1e-5f);
    if (tid == 0) invs[t] = 1.f / scale;
#pragma unroll
    for (int i = 0; i < 4; ++i) {
        float q = roundf(y[i] * nf * scale);
        q = fminf(fmaxf(q, -128.f), 127.f);
        q8[t * kD + tid + i * 256] = (int8_t)q;
    }
}

template <bool F16OUT>
__global__ __launch_bounds__(128) void bitgemm_lds(const int8_t* __restrict__ A8,
                                                   const int8_t* __restrict__ W8,
                                                   const float* __restrict__ invs,
                                                   const float* __restrict__ alpha,
                                                   const float* __restrict__ bias,
                                                   void* __restrict__ outp, int ostride) {
    __shared__ int8_t sw[2][64 * 64];
    const int tid  = threadIdx.x;
    const int lane = tid & 31, wv = tid >> 5;
    const int n  = lane & 15;
    const int hf = lane >> 4;
    const size_t trow  = (size_t)blockIdx.x * 256 + (size_t)wv * 64;
    const int    ocol0 = blockIdx.y * 64;

    const int8_t* arow[4];
#pragma unroll
    for (int at = 0; at < 4; ++at) arow[at] = A8 + (trow + at * 16 + n) * kD;

    v8i acc[4][4];
#pragma unroll
    for (int at = 0; at < 4; ++at)
#pragma unroll
        for (int bt = 0; bt < 4; ++bt) acc[at][bt] = {};

    stage_panel(W8, sw[0], ocol0, 0, tid);

    for (int kb = 0; kb < 16; ++kb) {
        const int buf = kb & 1;
        if (kb + 1 < 16) {
            stage_panel(W8, sw[buf ^ 1], ocol0, kb + 1, tid);
            wait_asynccnt<2>();
        } else {
            wait_asynccnt<0>();
        }
        __syncthreads();

        const int ab = kb * 64 + hf * 16;
        IFrag a[4];
#pragma unroll
        for (int at = 0; at < 4; ++at)
#pragma unroll
            for (int v = 0; v < 8; ++v)
                a[at].u[v] = *(const unsigned*)(arow[at] + ab + (v >> 2) * 32 + (v & 3) * 4);

#pragma unroll
        for (int bt = 0; bt < 4; ++bt) {
            IFrag b;
            const int8_t* lb = &sw[buf][(bt * 16 + n) * 64 + hf * 16];
#pragma unroll
            for (int v = 0; v < 8; ++v)
                b.u[v] = *(const unsigned*)(lb + (v >> 2) * 32 + (v & 3) * 4);
#pragma unroll
            for (int at = 0; at < 4; ++at)
                acc[at][bt] = __builtin_amdgcn_wmma_i32_16x16x64_iu8(
                    true, a[at].v, true, b.v, acc[at][bt], false, false);
        }
        __syncthreads();
    }

    const float al = alpha[0];
#pragma unroll
    for (int at = 0; at < 4; ++at) {
#pragma unroll
        for (int r = 0; r < 8; ++r) {
            const size_t row = trow + at * 16 + r + hf * 8;
            const float isc = invs[row] * al;
#pragma unroll
            for (int bt = 0; bt < 4; ++bt) {
                const int col = ocol0 + bt * 16 + n;
                const float y = (float)acc[at][bt][r] * isc + bias[col];
                if (F16OUT)
                    ((_Float16*)outp)[row * (size_t)ostride + col] = (_Float16)y;
                else
                    ((float*)outp)[row * (size_t)ostride + col] = y;
            }
        }
    }
}

__global__ __launch_bounds__(32) void gqa_attention(const _Float16* __restrict__ Qh,
                                                    const _Float16* __restrict__ Kh,
                                                    const _Float16* __restrict__ Vh,
                                                    float* __restrict__ O) {
    const int lane = threadIdx.x;
    const int n  = lane & 15;
    const int hf = lane >> 4;
    const int hb  = blockIdx.y;
    const int b   = hb >> 3, qh = hb & 7, kvh = qh >> 2;
    const size_t qrow0 = (size_t)b * kS + (size_t)blockIdx.x * 16;
    const _Float16* Qb = Qh + qrow0 * kD + qh * kHD;
    const _Float16* Kb = Kh + (size_t)b * kS * kKV + kvh * kHD;
    const _Float16* Vb = Vh + (size_t)b * kS * kKV + kvh * kHD;

    HFrag qa[4];
#pragma unroll
    for (int c = 0; c < 4; ++c)
#pragma unroll
        for (int v = 0; v < 8; ++v) {
            const int k = c * 32 + hf * 16 + v * 2;
            qa[c].u[v] = *(const unsigned*)(Qb + (size_t)n * kD + k);
        }

    v8f oacc[8];
#pragma unroll
    for (int t = 0; t < 8; ++t) oacc[t] = {};
    float mrun[8], lrun[8];
#pragma unroll
    for (int r = 0; r < 8; ++r) { mrun[r] = -__builtin_inff(); lrun[r] = 0.f; }

    __shared__ float sP[16][33];
    const float sm = 0.088388347648318447f;

    for (int j = 0; j < kS / 32; ++j) {
        const int key0 = j * 32;
        v8f s0 = {}, s1 = {};
#pragma unroll
        for (int c = 0; c < 4; ++c) {
            HFrag k0f, k1f;
#pragma unroll
            for (int v = 0; v < 8; ++v) {
                const int d = c * 32 + hf * 16 + v * 2;
                k0f.u[v] = *(const unsigned*)(Kb + (size_t)(key0 + n) * kKV + d);
                k1f.u[v] = *(const unsigned*)(Kb + (size_t)(key0 + 16 + n) * kKV + d);
            }
            s0 = __builtin_amdgcn_wmma_f32_16x16x32_f16(false, qa[c].v, false, k0f.v,
                                                        (short)0, s0, false, false);
            s1 = __builtin_amdgcn_wmma_f32_16x16x32_f16(false, qa[c].v, false, k1f.v,
                                                        (short)0, s1, false, false);
        }
        float corr[8];
#pragma unroll
        for (int r = 0; r < 8; ++r) {
            const float a0 = s0[r] * sm, a1 = s1[r] * sm;
            float mr = fmaxf(a0, a1);
            mr = fmaxf(mr, __shfl_xor(mr, 1, 32));
            mr = fmaxf(mr, __shfl_xor(mr, 2, 32));
            mr = fmaxf(mr, __shfl_xor(mr, 4, 32));
            mr = fmaxf(mr, __shfl_xor(mr, 8, 32));
            const float mnew = fmaxf(mrun[r], mr);
            corr[r] = __expf(mrun[r] - mnew);
            const float p0 = __expf(a0 - mnew);
            const float p1 = __expf(a1 - mnew);
            float rs = p0 + p1;
            rs += __shfl_xor(rs, 1, 32);
            rs += __shfl_xor(rs, 2, 32);
            rs += __shfl_xor(rs, 4, 32);
            rs += __shfl_xor(rs, 8, 32);
            lrun[r] = lrun[r] * corr[r] + rs;
            mrun[r] = mnew;
            sP[r + hf * 8][n]      = p0;
            sP[r + hf * 8][16 + n] = p1;
        }
        __syncthreads();
        HFrag pa;
#pragma unroll
        for (int v = 0; v < 8; ++v) {
            const int k = hf * 16 + v * 2;
            pa.h[2 * v]     = (_Float16)sP[n][k];
            pa.h[2 * v + 1] = (_Float16)sP[n][k + 1];
        }
        __syncthreads();
#pragma unroll
        for (int t = 0; t < 8; ++t) {
            v8f o = oacc[t];
#pragma unroll
            for (int r = 0; r < 8; ++r) o[r] *= corr[r];
            HFrag vbf;
#pragma unroll
            for (int v = 0; v < 8; ++v) {
                vbf.h[2 * v]     = Vb[(size_t)(key0 + hf * 16 + 2 * v)     * kKV + t * 16 + n];
                vbf.h[2 * v + 1] = Vb[(size_t)(key0 + hf * 16 + 2 * v + 1) * kKV + t * 16 + n];
            }
            oacc[t] = __builtin_amdgcn_wmma_f32_16x16x32_f16(false, pa.v, false, vbf.v,
                                                             (short)0, o, false, false);
        }
    }

    float* Orow = O + qrow0 * kD + qh * kHD;
#pragma unroll
    for (int r = 0; r < 8; ++r) {
        const float linv = 1.f / lrun[r];
#pragma unroll
        for (int t = 0; t < 8; ++t)
            Orow[(size_t)(r + hf * 8) * kD + t * 16 + n] = oacc[t][r] * linv;
    }
}

extern "C" void kernel_launch(void* const* d_in, const int* in_sizes, int n_in,
                              void* d_out, int out_size, void* d_ws, size_t ws_size,
                              hipStream_t stream) {
    (void)in_sizes; (void)n_in; (void)out_size; (void)ws_size;
    const float* query = (const float*)d_in[0];
    const float* key   = (const float*)d_in[1];
    const float* value = (const float*)d_in[2];
    const float* q_w   = (const float*)d_in[3];
    const float* q_b   = (const float*)d_in[4];
    const float* k_w   = (const float*)d_in[5];
    const float* k_b   = (const float*)d_in[6];
    const float* v_w   = (const float*)d_in[7];
    const float* v_b   = (const float*)d_in[8];
    const float* ln_g  = (const float*)d_in[9];
    const float* ln_b  = (const float*)d_in[10];
    const float* out_w = (const float*)d_in[11];
    const float* out_b = (const float*)d_in[12];

    uint8_t* base = (uint8_t*)d_ws;
    size_t off = 0;
    auto take = [&](size_t bytes) -> void* {
        off = (off + 255) & ~(size_t)255;
        void* p = base + off;
        off += bytes;
        return p;
    };
    float*  partials = (float*)take(4 * 256 * sizeof(float));
    float*  alphas   = (float*)take(4 * sizeof(float));
    int8_t* sgnQ = (int8_t*)take((size_t)kD * kD);
    int8_t* sgnK = (int8_t*)take((size_t)kKV * kD);
    int8_t* sgnV = (int8_t*)take((size_t)kKV * kD);
    int8_t* sgnO = (int8_t*)take((size_t)kD * kD);
    int8_t* aq8  = (int8_t*)take((size_t)kT * kD);
    int8_t* ak8  = (int8_t*)take((size_t)kT * kD);
    int8_t* av8  = (int8_t*)take((size_t)kT * kD);
    int8_t* ao8  = (int8_t*)take((size_t)kT * kD);
    float*  ivq  = (float*)take(kT * sizeof(float));
    float*  ivk  = (float*)take(kT * sizeof(float));
    float*  ivv  = (float*)take(kT * sizeof(float));
    float*  ivo  = (float*)take(kT * sizeof(float));
    _Float16* Qh = (_Float16*)take((size_t)kT * kD * sizeof(_Float16));
    _Float16* Kh = (_Float16*)take((size_t)kT * kKV * sizeof(_Float16));
    _Float16* Vh = (_Float16*)take((size_t)kT * kKV * sizeof(_Float16));
    float*  attO = (float*)take((size_t)kT * kD * sizeof(float));

    weight_prep<<<256, 256, 0, stream>>>(q_w,   sgnQ, partials + 0 * 256, kD * kD);
    weight_prep<<<256, 256, 0, stream>>>(k_w,   sgnK, partials + 1 * 256, kKV * kD);
    weight_prep<<<256, 256, 0, stream>>>(v_w,   sgnV, partials + 2 * 256, kKV * kD);
    weight_prep<<<256, 256, 0, stream>>>(out_w, sgnO, partials + 3 * 256, kD * kD);
    alpha_reduce<<<4, 256, 0, stream>>>(partials, alphas);

    act_quant<<<kT, 256, 0, stream>>>(query, aq8, ivq);
    act_quant<<<kT, 256, 0, stream>>>(key,   ak8, ivk);
    act_quant<<<kT, 256, 0, stream>>>(value, av8, ivv);

    bitgemm_lds<true><<<dim3(kT / 256, kD / 64),  128, 0, stream>>>(aq8, sgnQ, ivq, alphas + 0, q_b, Qh, kD);
    bitgemm_lds<true><<<dim3(kT / 256, kKV / 64), 128, 0, stream>>>(ak8, sgnK, ivk, alphas + 1, k_b, Kh, kKV);
    bitgemm_lds<true><<<dim3(kT / 256, kKV / 64), 128, 0, stream>>>(av8, sgnV, ivv, alphas + 2, v_b, Vh, kKV);

    gqa_attention<<<dim3(kS / 16, kBt * kQH), 32, 0, stream>>>(Qh, Kh, Vh, attO);

    ln_act_quant<<<kT, 256, 0, stream>>>(attO, ln_g, ln_b, ao8, ivo);
    bitgemm_lds<false><<<dim3(kT / 256, kD / 64), 128, 0, stream>>>(ao8, sgnO, ivo, alphas + 3, out_b, d_out, kD);
}
